// TemporalSpectralAdapter_51299089383904
// MI455X (gfx1250) — compile-verified, matrix operand order checked by reading, not run on hardware
//
#include <hip/hip_runtime.h>
#include <math.h>

// ---------------- problem dims ----------------
#define NB    16     // batch
#define CC    256    // channels
#define SS    64     // sequence
#define HH    16
#define WW    11
#define HWW   176    // h*w
#define PBB   4      // part blocks
#define FBINS 33     // rfft bins of 64
#define LOWF  8      // low-frequency cut
#define HIDN  64     // mlp hidden
#define NBR   5      // branches

#define CDIV(a,b) (((a)+(b)-1)/(b))

typedef __attribute__((ext_vector_type(16))) __bf16 v16bf;
typedef __attribute__((ext_vector_type(8)))  float  v8f;
typedef int vint4 __attribute__((vector_size(16)));   // matches async-LDS builtin param type

// gfx1250 async global->LDS path (guarded; falls back to plain vector loads)
#if defined(__gfx1250__) && __has_builtin(__builtin_amdgcn_global_load_async_to_lds_b128) && \
    __has_builtin(__builtin_amdgcn_s_wait_asynccnt)
#define USE_ASYNC_LDS 1
#else
#define USE_ASYNC_LDS 0
#endif

__device__ __forceinline__ float gelu_f(float x) {
    return 0.5f * x * (1.f + erff(x * 0.7071067811865476f));
}

// epilogue codes for the generic WMMA GEMM
#define EPI_NONE       0
#define EPI_BIAS       1
#define EPI_BIAS_GELU  2
#define EPI_BN_GELU    3   // gelu(v * g[n]*rsqrt(1+eps) + b[n])
#define EPI_BN_SIGMOID 4   // sigmoid(v * g[n]*rsqrt(1+eps) + b[n])

__device__ __forceinline__ v16bf pack16(const float4& a0, const float4& a1,
                                        const float4& a2, const float4& a3)
{
    v16bf v;
    v[0]  = (__bf16)a0.x; v[1]  = (__bf16)a0.y; v[2]  = (__bf16)a0.z; v[3]  = (__bf16)a0.w;
    v[4]  = (__bf16)a1.x; v[5]  = (__bf16)a1.y; v[6]  = (__bf16)a1.z; v[7]  = (__bf16)a1.w;
    v[8]  = (__bf16)a2.x; v[9]  = (__bf16)a2.y; v[10] = (__bf16)a2.z; v[11] = (__bf16)a2.w;
    v[12] = (__bf16)a3.x; v[13] = (__bf16)a3.y; v[14] = (__bf16)a3.z; v[15] = (__bf16)a3.w;
    return v;
}

// =====================================================================
// Generic WMMA GEMM: D[M,N] = epi( A[M,K] @ B^T ),  B given as (N,K) row-major
// One wave computes a 16x16 tile via v_wmma_f32_16x16x32_bf16.
// A rows can be remapped (FFT-bin slices): element (r,k) lives at
//   A[(r/rows_per_grp)*grp_stride + (row_off + r%rows_per_grp)*lda + k]
// M % 16 == 0, K % 32 == 0; BOUND instantiation bounds-checks N.
// All fragment loads are aligned float4 (K and kb offsets are mult. of 8).
// =====================================================================
template <bool BOUND>
__global__ __launch_bounds__(32)
void gemm_wmma_kernel(const float* __restrict__ A, const float* __restrict__ B,
                      const float* __restrict__ g, const float* __restrict__ bias,
                      float* __restrict__ D,
                      int M, int N, int K,
                      int rows_per_grp, long grp_stride, int row_off, int lda, int epi)
{
    int lane = threadIdx.x & 31;
    int tile_n = blockIdx.x, tile_m = blockIdx.y;
    int half = lane >> 4;         // 0: lanes 0-15, 1: lanes 16-31
    int lr   = lane & 15;
    int m  = tile_m * 16 + lr;    // A-row owned by this lane
    int nn = tile_n * 16 + lr;    // B-row (output col) owned by this lane
    long abase = (long)(m / rows_per_grp) * grp_stride
               + (long)(row_off + (m % rows_per_grp)) * (long)lda;
    long bbase = (long)nn * K;
    bool nv = !BOUND || (nn < N);

    v8f acc = {};
#pragma unroll 2
    for (int kk = 0; kk < K; kk += 32) {
        // 16-bit A-matrix 16x32 layout (ISA 7.12.2):
        //   elems 0..7  -> K = kk + half*8 + j ; elems 8..15 -> +16
        int kb0 = kk + half * 8;
        int kb1 = kk + 16 + half * 8;
        float4 a0 = *(const float4*)(A + abase + kb0);
        float4 a1 = *(const float4*)(A + abase + kb0 + 4);
        float4 a2 = *(const float4*)(A + abase + kb1);
        float4 a3 = *(const float4*)(A + abase + kb1 + 4);
        float4 z = {0.f, 0.f, 0.f, 0.f};
        float4 b0 = z, b1 = z, b2 = z, b3 = z;
        if (nv) {
            b0 = *(const float4*)(B + bbase + kb0);
            b1 = *(const float4*)(B + bbase + kb0 + 4);
            b2 = *(const float4*)(B + bbase + kb1);
            b3 = *(const float4*)(B + bbase + kb1 + 4);
        }
        v16bf va = pack16(a0, a1, a2, a3);
        v16bf vb = pack16(b0, b1, b2, b3);
        acc = __builtin_amdgcn_wmma_f32_16x16x32_bf16(
            /*neg_a=*/false, va, /*neg_b=*/false, vb,
            /*c_mod=*/(short)0, acc, /*reuse_a=*/false, /*reuse_b=*/false);
    }

    if (nv) {
        float sc = 1.f, bs = 0.f;
        if (epi == EPI_BN_GELU || epi == EPI_BN_SIGMOID) {
            sc = g[nn] * rsqrtf(1.f + 1e-5f);
            bs = bias[nn];
        } else if (epi == EPI_BIAS || epi == EPI_BIAS_GELU) {
            bs = bias ? bias[nn] : 0.f;
        }
#pragma unroll
        for (int r = 0; r < 8; ++r) {
            int mr = tile_m * 16 + half * 8 + r;   // f32 C/D: VGPR r -> M = r (+8 hi half)
            float v = acc[r];
            if (epi == EPI_BN_GELU || epi == EPI_BN_SIGMOID) v = v * sc + bs; else v += bs;
            if (epi == EPI_BIAS_GELU || epi == EPI_BN_GELU)  v = gelu_f(v);
            else if (epi == EPI_BN_SIGMOID)                  v = 1.f / (1.f + expf(-v));
            D[(long)mr * N + nn] = v;
        }
    }
}

// ============ streaming reductions over x ============

// pd[n,pb,s,c] = mean over w and the 4 h's of group pb  (44 contiguous floats)
__global__ __launch_bounds__(256)
void reduce_pd_kernel(const float* __restrict__ x, float* __restrict__ pd)
{
    long idx = (long)blockIdx.x * blockDim.x + threadIdx.x;
    long total = (long)NB * CC * SS * PBB;
    if (idx >= total) return;
    int pb = (int)(idx & 3);
    long r = idx >> 2;                 // (n*C + c)*S + s
    int s = (int)(r % SS);
    long nc = r / SS;
    int c = (int)(nc % CC);
    int n = (int)(nc / CC);
    const float4* p4 = (const float4*)(x + r * HWW + (long)pb * 44);
    float sum = 0.f;
#pragma unroll
    for (int i = 0; i < 11; ++i) { float4 v = p4[i]; sum += v.x + v.y + v.z + v.w; }
    pd[(((long)n * PBB + pb) * SS + s) * CC + c] = sum * (1.f / 44.f);
}

// t_in[n,s,c] = mean over pb of pd  (== desc transposed, pre-LayerNorm)
__global__ __launch_bounds__(256)
void pbmean_kernel(const float* __restrict__ pd, float* __restrict__ out)
{
    long idx = (long)blockIdx.x * blockDim.x + threadIdx.x;
    long total = (long)NB * SS * CC;
    if (idx >= total) return;
    int c = (int)(idx % CC);
    int s = (int)((idx / CC) % SS);
    int n = (int)(idx / ((long)CC * SS));
    float sum = 0.f;
#pragma unroll
    for (int pb = 0; pb < PBB; ++pb)
        sum += pd[(((long)n * PBB + pb) * SS + s) * CC + c];
    out[idx] = sum * 0.25f;
}

// xm2r[n,hw,c] = mean over s of x  (x.mean(axis=2), stored (n,h*w,c))
__global__ __launch_bounds__(256)
void xmean_s_kernel(const float* __restrict__ x, float* __restrict__ out)
{
    long idx = (long)blockIdx.x * blockDim.x + threadIdx.x;   // ((n*C + c)*HW + hw)
    long total = (long)NB * CC * HWW;
    if (idx >= total) return;
    int hw = (int)(idx % HWW);
    long nc = idx / HWW;
    int c = (int)(nc % CC);
    int n = (int)(nc / CC);
    float sum = 0.f;
    const float* base = x + (nc * SS) * HWW + hw;
    for (int s = 0; s < SS; ++s) sum += base[(long)s * HWW];
    out[((long)n * HWW + hw) * CC + c] = sum * (1.f / 64.f);
}

// LayerNorm over C=256 per row (one 256-thread block per row)
__global__ __launch_bounds__(256)
void ln_kernel(const float* __restrict__ in, float* __restrict__ out,
               const float* __restrict__ g, const float* __restrict__ b)
{
    __shared__ float red[256];
    __shared__ float s_mean, s_rstd;
    long row = blockIdx.x;
    int c = threadIdx.x;
    float v = in[row * CC + c];
    red[c] = v; __syncthreads();
    for (int o = 128; o > 0; o >>= 1) { if (c < o) red[c] += red[c + o]; __syncthreads(); }
    if (c == 0) s_mean = red[0] * (1.f / 256.f);
    __syncthreads();
    float d = v - s_mean;
    red[c] = d * d; __syncthreads();
    for (int o = 128; o > 0; o >>= 1) { if (c < o) red[c] += red[c + o]; __syncthreads(); }
    if (c == 0) s_rstd = rsqrtf(red[0] * (1.f / 256.f) + 1e-5f);
    __syncthreads();
    out[row * CC + c] = d * s_rstd * g[c] + b[c];
}

// ============ spectral (direct DFT, 64-pt, ortho) ============

__global__ __launch_bounds__(256)
void dft_fwd_kernel(const float* __restrict__ in, float* __restrict__ amp,
                    float* __restrict__ ph, int batch)
{
    long idx = (long)blockIdx.x * blockDim.x + threadIdx.x;
    long total = (long)batch * FBINS * CC;
    if (idx >= total) return;
    int c = (int)(idx % CC);
    long rk = idx / CC;
    int k = (int)(rk % FBINS);
    int n = (int)(rk / FBINS);
    float w0 = -6.283185307179586f / 64.f * (float)k;
    float re = 0.f, im = 0.f;
    for (int t = 0; t < SS; ++t) {
        float xv = in[((long)n * SS + t) * CC + c];
        float sn, cs;
        __sincosf(w0 * (float)t, &sn, &cs);
        re += xv * cs; im += xv * sn;
    }
    amp[idx] = 0.125f * sqrtf(re * re + im * im);
    ph[idx]  = atan2f(im, re);
}

// irfft of {low bins -> idn, high bins -> dyn}. idn may be null (part branch).
__global__ __launch_bounds__(256)
void idft_kernel(const float* __restrict__ am, const float* __restrict__ pm,
                 float* __restrict__ dyn, float* __restrict__ idn, int batch)
{
    long idx = (long)blockIdx.x * blockDim.x + threadIdx.x;
    long total = (long)batch * SS * CC;
    if (idx >= total) return;
    int c = (int)(idx % CC);
    int t = (int)((idx / CC) % SS);
    int n = (int)(idx / ((long)CC * SS));
    const float w0 = 6.283185307179586f / 64.f;
    long base = ((long)n * FBINS) * CC + c;
    float sid, sdy = 0.f;
    { float a = am[base], p = pm[base]; sid = a * cosf(p); }      // bin 0 (Re only)
    for (int k = 1; k < LOWF; ++k) {
        float a = am[base + (long)k * CC], p = pm[base + (long)k * CC];
        sid += 2.f * a * cosf(p + w0 * (float)(k * t));
    }
    for (int k = LOWF; k < 32; ++k) {
        float a = am[base + (long)k * CC], p = pm[base + (long)k * CC];
        sdy += 2.f * a * cosf(p + w0 * (float)(k * t));
    }
    { float a = am[base + 32l * CC], p = pm[base + 32l * CC];     // Nyquist (Re only)
      sdy += ((t & 1) ? -1.f : 1.f) * a * cosf(p); }
    dyn[idx] = 0.125f * sdy;
    if (idn) idn[idx] = 0.125f * sid;
}

// amp' = amp*(1+gs*tanh(mlp)) or ph' = ph + gs*tanh(mlp); rows remap into (n,33,c)
__global__ __launch_bounds__(256)
void modulate_kernel(const float* __restrict__ src, const float* __restrict__ mout,
                     float* __restrict__ dst, int M, int F, int k0, float gs, int addmode)
{
    long idx = (long)blockIdx.x * blockDim.x + threadIdx.x;
    if (idx >= (long)M * CC) return;
    int c = (int)(idx % CC);
    int r = (int)(idx / CC);
    int n = r / F, k = r % F;
    long o = ((long)n * FBINS + k0 + k) * CC + c;
    float t = tanhf(mout[idx]);
    dst[o] = addmode ? (src[o] + gs * t) : (src[o] * (1.f + gs * t));
}

// ============ conv-as-GEMM prep ============

// im2col: A_aug[(n,s)][tap*256+c] = tokens[n, s+shift(tap), c] (0-padded); taps 0-2: w3, 3-7: w5
__global__ __launch_bounds__(256)
void im2col_kernel(const float* __restrict__ tokens, float* __restrict__ out)
{
    long idx = (long)blockIdx.x * blockDim.x + threadIdx.x;
    long total = (long)NB * SS * 8 * CC;
    if (idx >= total) return;
    int k = (int)(idx % (8 * CC));
    long row = idx / (8 * CC);
    int n = (int)(row / SS), s = (int)(row % SS);
    int tap = k / CC, c = k % CC;
    int shift = (tap < 3) ? (tap - 1) : (tap - 5);
    int ss = s + shift;
    float v = (ss >= 0 && ss < SS) ? tokens[((long)n * SS + ss) * CC + c] : 0.f;
    out[idx] = v;
}

// conv weights packed directly in (N,K) = (o, tap*256+c) layout for the GEMM
__global__ __launch_bounds__(256)
void convpack_kernel(const float* __restrict__ w3, const float* __restrict__ w5,
                     float* __restrict__ out)
{
    long idx = (long)blockIdx.x * blockDim.x + threadIdx.x;   // o*2048 + k
    long total = (long)CC * 8 * CC;
    if (idx >= total) return;
    int k = (int)(idx % (8 * CC));
    int o = (int)(idx / (8 * CC));
    int tap = k / CC, c = k % CC;
    float v = (tap < 3) ? 0.5f * w3[((long)o * CC + c) * 3 + tap]
                        : 0.5f * w5[((long)o * CC + c) * 5 + (tap - 3)];
    out[idx] = v;
}

// small weight transpose: out[c][r] = in[r][c]   (to give every GEMM (N,K) B-layout)
__global__ __launch_bounds__(256)
void transpose_kernel(const float* __restrict__ in, float* __restrict__ out,
                      int rows, int cols)
{
    long idx = (long)blockIdx.x * blockDim.x + threadIdx.x;
    if (idx >= (long)rows * cols) return;
    int c = (int)(idx % cols);
    int r = (int)(idx / cols);
    out[(long)c * rows + r] = in[idx];
}

// ============ small glue kernels ============

__global__ __launch_bounds__(256)
void mean_s_kernel(const float* __restrict__ in, float* __restrict__ out, int batch)
{   // out[b,c] = mean over s of in[b,s,c]
    long idx = (long)blockIdx.x * blockDim.x + threadIdx.x;
    if (idx >= (long)batch * CC) return;
    int c = (int)(idx % CC);
    int b = (int)(idx / CC);
    float sum = 0.f;
    for (int s = 0; s < SS; ++s) sum += in[((long)b * SS + s) * CC + c];
    out[idx] = sum * (1.f / 64.f);
}

__global__ __launch_bounds__(256)
void diff_kernel(const float* __restrict__ a, const float* __restrict__ b,
                 float* __restrict__ o, long total)
{
    long idx = (long)blockIdx.x * blockDim.x + threadIdx.x;
    if (idx < total) o[idx] = a[idx] - b[idx];
}

// pg16[n,c,h] = linear interp of pgs[n,pb,c] over pb (L=4 -> 16)
__global__ __launch_bounds__(256)
void interp_kernel(const float* __restrict__ pgs, float* __restrict__ pg16)
{
    long idx = (long)blockIdx.x * blockDim.x + threadIdx.x;
    long total = (long)NB * CC * HH;
    if (idx >= total) return;
    int h = (int)(idx % HH);
    int c = (int)((idx / HH) % CC);
    int n = (int)(idx / ((long)HH * CC));
    float ci = fminf(fmaxf(0.25f * (float)h - 0.375f, 0.f), 3.f);
    int i0 = (int)floorf(ci);
    int i1 = min(i0 + 1, 3);
    float w = ci - (float)i0;
    float v0 = pgs[((long)n * PBB + i0) * CC + c];
    float v1 = pgs[((long)n * PBB + i1) * CC + c];
    pg16[idx] = v0 * (1.f - w) + v1 * w;
}

__global__ void softmax5_kernel(const float* __restrict__ logits, float* __restrict__ bw)
{
    int n = threadIdx.x;
    if (n >= NB) return;
    float l[NBR], mx = -1e30f;
#pragma unroll
    for (int j = 0; j < NBR; ++j) { l[j] = logits[n * NBR + j]; mx = fmaxf(mx, l[j]); }
    float sum = 0.f;
#pragma unroll
    for (int j = 0; j < NBR; ++j) { l[j] = expf(l[j] - mx); sum += l[j]; }
#pragma unroll
    for (int j = 0; j < NBR; ++j) bw[n * NBR + j] = l[j] / sum;
}

__global__ __launch_bounds__(256)
void branchmix_kernel(const float* __restrict__ idnp, const float* __restrict__ local,
                      const float* __restrict__ dynp, const float* __restrict__ gap,
                      const float* __restrict__ pctx, const float* __restrict__ bw,
                      float* __restrict__ d0)
{
    long idx = (long)blockIdx.x * blockDim.x + threadIdx.x;
    long total = (long)NB * SS * CC;
    if (idx >= total) return;
    int n = (int)(idx / ((long)SS * CC));
    const float* w = bw + n * NBR;
    d0[idx] = w[0] * idnp[idx] + w[1] * local[idx] + w[2] * dynp[idx]
            + w[3] * gap[idx]  + w[4] * pctx[idx];
}

// out = x + rs[c]*delta[n,s,c] * gate2d[n,hw,c] * pg16[n,c,h]   (the HBM pass)
// x is streamed through the gfx1250 async global->LDS pipe when available:
// 16B per lane to a private LDS slot, overlap gate-index math, s_wait_asynccnt 0.
// Grid is sized exactly (total/4 % 256 == 0), so EXEC is all-ones throughout.
__global__ __launch_bounds__(256)
void final_kernel(const float* __restrict__ x, const float* __restrict__ delta,
                  const float* __restrict__ gate2d, const float* __restrict__ pg16,
                  const float* __restrict__ rs, float* __restrict__ out)
{
#if USE_ASYNC_LDS
    __shared__ float tile[256 * 4];
#endif
    long gid = (long)blockIdx.x * blockDim.x + threadIdx.x;
    long e = gid * 4;
#if USE_ASYNC_LDS
    __builtin_amdgcn_global_load_async_to_lds_b128(
        (__attribute__((address_space(1))) vint4*)(x + e),
        (__attribute__((address_space(3))) vint4*)&tile[threadIdx.x * 4], 0, 0);
#else
    float4 xv = *(const float4*)(x + e);
#endif
    long row = e / HWW;               // (n*C + c)*S + s  (176 % 4 == 0 -> one row/thread)
    int loc = (int)(e % HWW);
    int s = (int)(row % SS);
    long nc = row / SS;
    int c = (int)(nc % CC);
    int n = (int)(nc / CC);
    float dval = rs[c] * delta[((long)n * SS + s) * CC + c];
    float gg[4];
#pragma unroll
    for (int j = 0; j < 4; ++j) {
        int l = loc + j;
        float g2 = gate2d[((long)n * HWW + l) * CC + c];
        float pg = pg16[((long)n * CC + c) * HH + (l / WW)];
        gg[j] = dval * g2 * pg;
    }
#if USE_ASYNC_LDS
    __builtin_amdgcn_s_wait_asynccnt(0);
    float4 xv = *(const float4*)&tile[threadIdx.x * 4];
#endif
    float4 ov = { xv.x + gg[0], xv.y + gg[1], xv.z + gg[2], xv.w + gg[3] };
    *(float4*)(out + e) = ov;
}

// =====================================================================
// host side
// =====================================================================
static inline void gemm(hipStream_t st, const float* A, const float* B, const float* g,
                        const float* bias, float* D, int M, int N, int K,
                        int rpg, long gstr, int roff, int lda, int epi)
{
    dim3 grid(CDIV(N, 16), M / 16);
    if ((N & 15) == 0)
        gemm_wmma_kernel<false><<<grid, 32, 0, st>>>(A, B, g, bias, D, M, N, K,
                                                     rpg, gstr, roff, lda, epi);
    else
        gemm_wmma_kernel<true><<<grid, 32, 0, st>>>(A, B, g, bias, D, M, N, K,
                                                    rpg, gstr, roff, lda, epi);
}

// Input ordering assumption: d_in[0] = x, then the `params` pytree leaves in
// jax tree-flatten order (dict keys sorted alphabetically, recursively).
enum {
    IN_X = 0,
    IN_AMP_L1B, IN_AMP_L1W, IN_AMP_L2B, IN_AMP_L2W,      // amp_mlp
    IN_BOUT_B, IN_BOUT_G, IN_BOUT_W,                      // branch_out_proj
    IN_BSRC_B, IN_BSRC_G, IN_BSRC_W,                      // branch_source_proj
    IN_BSUM_L1B, IN_BSUM_L1W, IN_BSUM_L2B, IN_BSUM_L2W,  // branch_summary
    IN_IDP_B, IN_IDP_G, IN_IDP_W,                         // identity_proj
    IN_LF_B, IN_LF_G, IN_LF_W,                            // local_fuse
    IN_W3, IN_W5,                                         // local_w3, local_w5
    IN_LAMP_L1B, IN_LAMP_L1W, IN_LAMP_L2B, IN_LAMP_L2W,  // low_amp_mlp
    IN_LPH_L1B, IN_LPH_L1W, IN_LPH_L2B, IN_LPH_L2W,      // low_phase_mlp
    IN_NORM_B, IN_NORM_G,                                 // norm
    IN_PAMP_L1B, IN_PAMP_L1W, IN_PAMP_L2B, IN_PAMP_L2W,  // part_amp_mlp
    IN_PNORM_B, IN_PNORM_G,                               // part_norm
    IN_PPROJ_B, IN_PPROJ_G, IN_PPROJ_W,                   // part_proj
    IN_PSG_B, IN_PSG_G, IN_PSG_W,                         // part_spatial_gate
    IN_PH_L1B, IN_PH_L1W, IN_PH_L2B, IN_PH_L2W,          // phase_mlp
    IN_RESSCALE,                                          // res_scale
    IN_SG_B, IN_SG_G, IN_SG_W,                            // spatial_gate
    IN_SPEC_B, IN_SPEC_G, IN_SPEC_W                       // spectral_proj
};

extern "C" void kernel_launch(void* const* d_in, const int* in_sizes, int n_in,
                              void* d_out, int out_size, void* d_ws, size_t ws_size,
                              hipStream_t stream)
{
    (void)in_sizes; (void)n_in; (void)out_size; (void)ws_size;
    const float* x = (const float*)d_in[IN_X];
    auto P = [&](int i) { return (const float*)d_in[i]; };

    const long SZ_MAP   = (long)NB * SS * CC;          // 262144
    const long SZ_PART  = (long)NB * PBB * SS * CC;    // 1048576
    const long SZ_SPEC  = (long)NB * FBINS * CC;       // 135168
    const long SZ_PSPEC = (long)NB * PBB * FBINS * CC; // 540672

    float* ws = (float*)d_ws;
    long cur = 0;
    auto alloc = [&](long n) { float* p = ws + cur; cur += n; return p; };

    float* pd     = alloc(SZ_PART);    // (n,pb,s,c)
    float* tin    = alloc(SZ_MAP);     // (n,s,c)  pre-LN tokens
    float* tokens = alloc(SZ_MAP);
    float* pt     = alloc(SZ_PART);
    float* amp    = alloc(SZ_SPEC);
    float* phs    = alloc(SZ_SPEC);
    float* ampm   = alloc(SZ_SPEC);
    float* phm    = alloc(SZ_SPEC);
    float* pamp   = alloc(SZ_PSPEC);
    float* pph    = alloc(SZ_PSPEC);
    float* pampm  = alloc(SZ_PSPEC);
    float* hid    = alloc((long)NB * PBB * 25 * HIDN); // max MLP rows = 1600
    float* mout   = alloc((long)NB * PBB * 25 * CC);
    float* i2c    = alloc((long)NB * SS * 8 * CC);     // 1024 x 2048
    float* cw     = alloc((long)CC * 8 * CC);          // (N,K) = 256 x 2048
    float* lc     = alloc(SZ_MAP);
    float* local  = alloc(SZ_MAP);
    float* dyn    = alloc(SZ_MAP);
    float* idn    = alloc(SZ_MAP);
    float* dynp   = alloc(SZ_MAP);
    float* idnp   = alloc(SZ_MAP);
    float* pdyn   = alloc(SZ_PART);
    float* pm1    = alloc(SZ_MAP);
    float* pctx   = alloc(SZ_MAP);
    float* pm2    = alloc((long)NB * PBB * CC);
    float* pgs    = alloc((long)NB * PBB * CC);
    float* pg16   = alloc((long)NB * CC * HH);
    float* gapin  = alloc(SZ_MAP);
    float* gap    = alloc(SZ_MAP);
    float* q      = alloc((long)NB * CC);
    float* shid   = alloc((long)NB * HIDN);
    float* slog   = alloc(128);
    float* bw     = alloc(128);
    float* d0     = alloc(SZ_MAP);
    float* delta  = alloc(SZ_MAP);
    float* xm2    = alloc((long)NB * HWW * CC);
    float* g2d    = alloc((long)NB * HWW * CC);
    // transposed small weights: everything becomes (N,K) B-layout for the GEMM
    float* wt_lamp1 = alloc(HIDN * CC);  float* wt_lamp2 = alloc(CC * HIDN);
    float* wt_amp1  = alloc(HIDN * CC);  float* wt_amp2  = alloc(CC * HIDN);
    float* wt_lph1  = alloc(HIDN * CC);  float* wt_lph2  = alloc(CC * HIDN);
    float* wt_ph1   = alloc(HIDN * CC);  float* wt_ph2   = alloc(CC * HIDN);
    float* wt_pamp1 = alloc(HIDN * CC);  float* wt_pamp2 = alloc(CC * HIDN);
    float* wt_bs1   = alloc(HIDN * CC);  float* wt_bs2   = alloc(NBR * HIDN);

    auto tr = [&](int src, float* dst, int rows, int cols) {
        transpose_kernel<<<CDIV((long)rows * cols, 256), 256, 0, stream>>>(P(src), dst, rows, cols);
    };
    tr(IN_LAMP_L1W, wt_lamp1, CC, HIDN);  tr(IN_LAMP_L2W, wt_lamp2, HIDN, CC);
    tr(IN_AMP_L1W,  wt_amp1,  CC, HIDN);  tr(IN_AMP_L2W,  wt_amp2,  HIDN, CC);
    tr(IN_LPH_L1W,  wt_lph1,  CC, HIDN);  tr(IN_LPH_L2W,  wt_lph2,  HIDN, CC);
    tr(IN_PH_L1W,   wt_ph1,   CC, HIDN);  tr(IN_PH_L2W,   wt_ph2,   HIDN, CC);
    tr(IN_PAMP_L1W, wt_pamp1, CC, HIDN);  tr(IN_PAMP_L2W, wt_pamp2, HIDN, CC);
    tr(IN_BSUM_L1W, wt_bs1,   CC, HIDN);  tr(IN_BSUM_L2W, wt_bs2,   HIDN, NBR);

    // ---- streaming reductions of x ----
    reduce_pd_kernel<<<CDIV(NB * CC * SS * PBB, 256), 256, 0, stream>>>(x, pd);
    xmean_s_kernel<<<CDIV(NB * CC * HWW, 256), 256, 0, stream>>>(x, xm2);
    pbmean_kernel<<<CDIV(SZ_MAP, 256), 256, 0, stream>>>(pd, tin);

    // ---- layernorms ----
    ln_kernel<<<NB * SS, 256, 0, stream>>>(tin, tokens, P(IN_NORM_G), P(IN_NORM_B));
    ln_kernel<<<NB * PBB * SS, 256, 0, stream>>>(pd, pt, P(IN_PNORM_G), P(IN_PNORM_B));

    // ---- forward DFT (tokens + part) ----
    dft_fwd_kernel<<<CDIV(SZ_SPEC, 256), 256, 0, stream>>>(tokens, amp, phs, NB);
    dft_fwd_kernel<<<CDIV(SZ_PSPEC, 256), 256, 0, stream>>>(pt, pamp, pph, NB * PBB);

    // ---- amp/phase MLP modulation (WMMA GEMMs, rows remapped to bin slices) ----
    const long GST = (long)FBINS * CC;
    auto mlp_mod = [&](const float* src, const float* w1t, int l1b, const float* w2t, int l2b,
                       float* dst, int batch, int F, int k0, float gs, int addmode) {
        int M = batch * F;
        gemm(stream, src, w1t, nullptr, P(l1b), hid, M, HIDN, CC, F, GST, k0, CC, EPI_BIAS_GELU);
        gemm(stream, hid, w2t, nullptr, P(l2b), mout, M, CC, HIDN, M, 0, 0, HIDN, EPI_BIAS);
        modulate_kernel<<<CDIV((long)M * CC, 256), 256, 0, stream>>>(src, mout, dst, M, F, k0, gs, addmode);
    };
    mlp_mod(amp,  wt_lamp1, IN_LAMP_L1B, wt_lamp2, IN_LAMP_L2B, ampm,  NB,       LOWF, 0,    0.15f, 0);
    mlp_mod(amp,  wt_amp1,  IN_AMP_L1B,  wt_amp2,  IN_AMP_L2B,  ampm,  NB,       25,   LOWF, 0.30f, 0);
    mlp_mod(phs,  wt_lph1,  IN_LPH_L1B,  wt_lph2,  IN_LPH_L2B,  phm,   NB,       LOWF, 0,    0.15f, 1);
    mlp_mod(phs,  wt_ph1,   IN_PH_L1B,   wt_ph2,   IN_PH_L2B,   phm,   NB,       25,   LOWF, 0.30f, 1);
    mlp_mod(pamp, wt_lamp1, IN_LAMP_L1B, wt_lamp2, IN_LAMP_L2B, pampm, NB * PBB, LOWF, 0,    0.15f, 0);
    mlp_mod(pamp, wt_pamp1, IN_PAMP_L1B, wt_pamp2, IN_PAMP_L2B, pampm, NB * PBB, 25,   LOWF, 0.30f, 0);

    // ---- inverse DFT: dynamic / identity reconstructions ----
    idft_kernel<<<CDIV(SZ_MAP, 256), 256, 0, stream>>>(ampm, phm, dyn, idn, NB);
    idft_kernel<<<CDIV(SZ_PART, 256), 256, 0, stream>>>(pampm, pph, pdyn, nullptr, NB * PBB);

    // ---- local branch: conv1d(3)+conv1d(5) as one K=2048 WMMA GEMM, then fuse proj ----
    im2col_kernel<<<CDIV((long)NB * SS * 8 * CC, 256), 256, 0, stream>>>(tokens, i2c);
    convpack_kernel<<<CDIV((long)CC * 8 * CC, 256), 256, 0, stream>>>(P(IN_W3), P(IN_W5), cw);
    gemm(stream, i2c, cw, nullptr, nullptr, lc, NB * SS, CC, 8 * CC, NB * SS, 0, 0, 8 * CC, EPI_NONE);
    gemm(stream, lc, P(IN_LF_W), P(IN_LF_G), P(IN_LF_B), local, NB * SS, CC, CC, NB * SS, 0, 0, CC, EPI_BN_GELU);

    // ---- spectral / identity projections ----
    gemm(stream, dyn, P(IN_SPEC_W), P(IN_SPEC_G), P(IN_SPEC_B), dynp, NB * SS, CC, CC, NB * SS, 0, 0, CC, EPI_BN_GELU);
    gemm(stream, idn, P(IN_IDP_W),  P(IN_IDP_G),  P(IN_IDP_B),  idnp, NB * SS, CC, CC, NB * SS, 0, 0, CC, EPI_BN_GELU);

    // ---- part branch: context proj + spatial gate ----
    pbmean_kernel<<<CDIV(SZ_MAP, 256), 256, 0, stream>>>(pdyn, pm1);
    gemm(stream, pm1, P(IN_PPROJ_W), P(IN_PPROJ_G), P(IN_PPROJ_B), pctx, NB * SS, CC, CC, NB * SS, 0, 0, CC, EPI_BN_GELU);
    mean_s_kernel<<<CDIV((long)NB * PBB * CC, 256), 256, 0, stream>>>(pdyn, pm2, NB * PBB);
    gemm(stream, pm2, P(IN_PSG_W), P(IN_PSG_G), P(IN_PSG_B), pgs, NB * PBB, CC, CC, NB * PBB, 0, 0, CC, EPI_BN_SIGMOID);
    interp_kernel<<<CDIV((long)NB * CC * HH, 256), 256, 0, stream>>>(pgs, pg16);

    // ---- gap branch ----
    diff_kernel<<<CDIV(SZ_MAP, 256), 256, 0, stream>>>(dynp, idnp, gapin, SZ_MAP);
    gemm(stream, gapin, P(IN_BSRC_W), P(IN_BSRC_G), P(IN_BSRC_B), gap, NB * SS, CC, CC, NB * SS, 0, 0, CC, EPI_BN_GELU);

    // ---- branch summary -> softmax weights ----
    mean_s_kernel<<<CDIV((long)NB * CC, 256), 256, 0, stream>>>(tokens, q, NB);
    gemm(stream, q, wt_bs1, nullptr, P(IN_BSUM_L1B), shid, NB, HIDN, CC, NB, 0, 0, CC, EPI_BIAS_GELU);
    gemm(stream, shid, wt_bs2, nullptr, P(IN_BSUM_L2B), slog, NB, NBR, HIDN, NB, 0, 0, HIDN, EPI_BIAS);
    softmax5_kernel<<<1, 32, 0, stream>>>(slog, bw);

    // ---- weighted branch mix + output projection ----
    branchmix_kernel<<<CDIV(SZ_MAP, 256), 256, 0, stream>>>(idnp, local, dynp, gap, pctx, bw, d0);
    gemm(stream, d0, P(IN_BOUT_W), P(IN_BOUT_G), P(IN_BOUT_B), delta, NB * SS, CC, CC, NB * SS, 0, 0, CC, EPI_BN_GELU);

    // ---- 2D spatial gate ----
    gemm(stream, xm2, P(IN_SG_W), P(IN_SG_G), P(IN_SG_B), g2d, NB * HWW, CC, CC, NB * HWW, 0, 0, CC, EPI_BN_SIGMOID);

    // ---- fused residual update (the HBM-bound pass) ----
    long total4 = (long)NB * CC * SS * HWW / 4;   // 11,534,336 (% 256 == 0)
    final_kernel<<<CDIV(total4, 256), 256, 0, stream>>>(x, delta, g2d, pg16, P(IN_RESSCALE), (float*)d_out);
}
